// FRCNN_9869834846840
// MI455X (gfx1250) — matrix operand order repaired and checked by reading, not compiled, not run on hardware
//
#include <hip/hip_runtime.h>
#include <hip/hip_bf16.h>
#include <stdint.h>

#ifndef ATH_USE_TDM
#define ATH_USE_TDM 1     // set 0 to fall back to sync LDS staging
#endif

// ---------------------------------------------------------------------------
// Types for CDNA5 WMMA (wave32): v16bf = 16 x bf16 (8 VGPRs), v8f = 8 x f32.
// ---------------------------------------------------------------------------
typedef __attribute__((ext_vector_type(16))) __bf16    v16bf;
typedef __attribute__((ext_vector_type(8)))  float     v8f;
typedef __attribute__((ext_vector_type(4)))  uint32_t  u32x4;
typedef __attribute__((ext_vector_type(8)))  uint32_t  u32x8;
typedef __attribute__((ext_vector_type(4)))  float     f32x4;

union Frag16 {            // one 16x32 bf16 A-fragment or 32x16 bf16 B-fragment
  v16bf    v;
  uint32_t u[8];
  u32x4    q[2];
};

__device__ __forceinline__ uint16_t bf16_rne(float a) {
  uint32_t u = __float_as_uint(a);
  return (uint16_t)((u + 0x7FFFu + ((u >> 16) & 1u)) >> 16);
}
// f32 pair -> packed bf16 pair: round-to-nearest (+0x8000) then one v_perm_b32
// result bytes: [hi.b3, hi.b2, lo.b3, lo.b2]  (sel 7,6,3,2 over {S0=hi, S1=lo})
__device__ __forceinline__ uint32_t pk_bf16(float lo, float hi) {
  uint32_t ul = __float_as_uint(lo) + 0x8000u;
  uint32_t uh = __float_as_uint(hi) + 0x8000u;
  return __builtin_amdgcn_perm(uh, ul, 0x07060302u);
}

// ---------------------------------------------------------------------------
// Kernel 1: ROI max-pool (feature map 512x50x50, SIZE=7, SCALE=1/16).
// Writes A[256][25088] bf16 (row = roi, col = c*49 + ph*7 + pw).
// ---------------------------------------------------------------------------
__global__ void roi_pool_kernel(const float* __restrict__ feat,
                                const float* __restrict__ boxes,
                                uint16_t* __restrict__ A) {
  const int n = blockIdx.x;
  const float4 bx = ((const float4*)boxes)[n];
  const int x1 = (int)(bx.x * 0.0625f);
  const int y1 = (int)(bx.y * 0.0625f);
  const int x2 = (int)(bx.z * 0.0625f);
  const int y2 = (int)(bx.w * 0.0625f);

  int rlo = y1 < 0 ? 0 : (y1 > 49 ? 49 : y1);
  int rhi = y2 > 49 ? 49 : y2;
  int rn  = rhi - rlo + 1; if (rn < 1) rn = 1;
  int clo = x1 < 0 ? 0 : (x1 > 49 ? 49 : x1);
  int chi = x2 > 49 ? 49 : x2;
  int cn  = chi - clo + 1; if (cn < 1) cn = 1;

  int rs[7], re[7], cs[7], ce[7];
#pragma unroll
  for (int i = 0; i < 7; ++i) {
    rs[i] = rlo + (i * rn) / 7;
    re[i] = rlo + ((i + 1) * rn + 6) / 7;
    cs[i] = clo + (i * cn) / 7;
    ce[i] = clo + ((i + 1) * cn + 6) / 7;
  }

  for (int c = threadIdx.x; c < 512; c += blockDim.x) {
    const float* fc = feat + (size_t)c * 2500;
    uint16_t* out = A + (size_t)n * 25088 + c * 49;
    for (int ph = 0; ph < 7; ++ph) {
      for (int pw = 0; pw < 7; ++pw) {
        float m = -3.402823466e38f;
        for (int y = rs[ph]; y < re[ph]; ++y) {
          const float* row = fc + y * 50;
          for (int x = cs[pw]; x < ce[pw]; ++x)
            m = fmaxf(m, row[x]);
        }
        out[ph * 7 + pw] = bf16_rne(m);
      }
    }
  }
}

// ---------------------------------------------------------------------------
// TDM issue: load a 64-row x 64-col bf16 tile (row stride K elems) into LDS
// with 16B padding every 128B -> LDS row pitch 144B (= 72 bf16), which makes
// the later ds_load_b128 fragment reads bank-conflict-free (stride 36 banks).
// Descriptor per CDNA5 ISA.
// ---------------------------------------------------------------------------
#if ATH_USE_TDM
__device__ __forceinline__ void tdm_stage_a(const uint16_t* gsrc,
                                            uint32_t lds_byte_off, int K) {
  const uint64_t ga = (uint64_t)(uintptr_t)gsrc;
  u32x4 g0;
  g0[0] = 1u;                                   // count=1, user descriptor
  g0[1] = lds_byte_off;                         // lds_addr
  g0[2] = (uint32_t)ga;                         // global_addr[31:0]
  g0[3] = (uint32_t)(ga >> 32) | 0x80000000u;   // global_addr[56:32] | type=2
  u32x8 g1;
  g1[0] = (1u << 16)        // data_size = 2 bytes
        | (1u << 20)        // pad_enable
        | (4u << 22)        // pad_interval: 32 DWORDs (128B) between pads
        | (3u << 25);       // pad_amount:   4 DWORDs (16B)
  g1[1] = ((uint32_t)K & 0xFFFFu) << 16;                    // tensor_dim0 lo
  g1[2] = (((uint32_t)K >> 16) & 0xFFFFu) | (256u << 16);   // dim0 hi | dim1
  g1[3] = 64u << 16;                                        // tile_dim0 = 64
  g1[4] = 64u;                                              // tile_dim1 = 64
  g1[5] = (uint32_t)K;                                      // dim0_stride lo
  g1[6] = 0u;
  g1[7] = 0u;
  u32x4 gz = {0u, 0u, 0u, 0u};                  // unused 3D/4D groups (zeros)
  asm volatile("tensor_load_to_lds %0, %1, %2, %3"
               :: "s"(g0), "s"(g1), "s"(gz), "s"(gz) : "memory");
}
#endif

// ---------------------------------------------------------------------------
// Kernel 2/3: H = relu(A @ W^T + b), bf16 out.
// A: [M x K] bf16 row-major (staged in LDS per 64-wide k-chunk, TDM double
// buffered). W: [N x K] f32 row-major streamed once from HBM, converted to
// bf16 in-register (1x v_perm + 2 adds per pair). 8 waves/block, block tile
// 64M x 128N, wave tile 32x32 via 2x2 v_wmma_f32_16x16x32_bf16.
// Grid: (N/128, M/64). Requires M%64==0, N%128==0, K%64==0.
// ---------------------------------------------------------------------------
__global__ void __launch_bounds__(256)
gemm_relu_bf16_kernel(const uint16_t* __restrict__ A,
                      const float* __restrict__ W,
                      const float* __restrict__ bias,
                      uint16_t* __restrict__ H,
                      int K, int N) {
  __shared__ uint16_t sA[2][64][72];            // 144B row pitch, 18.4 KB

  const int tid  = threadIdx.x;
  const int lane = tid & 31;
  const int wave = tid >> 5;
  const int l15  = lane & 15;
  const int lhi  = lane >> 4;                   // 0 or 1
  const int mBlk = blockIdx.y * 64;
  const int mLoc = (wave & 1) * 32;             // wave's M offset inside block
  const int n0   = blockIdx.x * 128 + (wave >> 1) * 32;

  // B fragment bases: lane holds W row n (== B column n), 16 contiguous K
  const float* wrow[2];
  wrow[0] = W + (size_t)(n0 + l15) * K + lhi * 16;
  wrow[1] = wrow[0] + (size_t)16 * K;

  uint16_t* sbase = &sA[0][0][0];
  const uint32_t ldsBase  = (uint32_t)(uintptr_t)sbase;   // LDS byte offset
  const int      bufElems = 64 * 72;

  v8f acc[2][2] = {};
  const int nchunks = K >> 6;                   // 64-wide k-chunks

#if ATH_USE_TDM
  if (wave == 0)
    tdm_stage_a(A + (size_t)mBlk * K, ldsBase, K);
#endif

  for (int c = 0; c < nchunks; ++c) {
    const int buf = c & 1;

#if ATH_USE_TDM
    if (wave == 0) {
      if (c + 1 < nchunks) {
        tdm_stage_a(A + (size_t)mBlk * K + (c + 1) * 64,
                    ldsBase + (uint32_t)((buf ^ 1) * bufElems * 2), K);
        __builtin_amdgcn_s_wait_tensorcnt(1);   // chunk c complete (in-order)
      } else {
        __builtin_amdgcn_s_wait_tensorcnt(0);
      }
    }
    __syncthreads();
#else
    {  // sync fallback staging: 256 threads x 32B = 8KB chunk
      const int r  = tid >> 2;
      const int cb = (tid & 3) * 16;
      const uint16_t* g = A + (size_t)(mBlk + r) * K + c * 64 + cb;
      u32x4 v0 = *(const u32x4*)g;
      u32x4 v1 = *(const u32x4*)(g + 8);
      *(u32x4*)&sA[buf][r][cb]     = v0;
      *(u32x4*)&sA[buf][r][cb + 8] = v1;
      __syncthreads();
    }
#endif

    // stream-ahead prefetch of W (~4KB ahead, 2 lines per fragment row)
    __builtin_prefetch(wrow[0] + c * 64 + 1024, 0, 1);
    __builtin_prefetch(wrow[0] + c * 64 + 1056, 0, 1);
    __builtin_prefetch(wrow[1] + c * 64 + 1024, 0, 1);
    __builtin_prefetch(wrow[1] + c * 64 + 1056, 0, 1);

    const uint16_t* sb = sbase + buf * bufElems;
#pragma unroll
    for (int s = 0; s < 2; ++s) {               // two 32-wide WMMA k-steps
      const int kk = c * 64 + s * 32;
      Frag16 a[2], b[2];
#pragma unroll
      for (int mi = 0; mi < 2; ++mi) {          // A fragments from LDS
        const uint16_t* p = sb + (mLoc + mi * 16 + l15) * 72 + s * 32 + lhi * 16;
        a[mi].q[0] = *(const u32x4*)p;
        a[mi].q[1] = *(const u32x4*)(p + 8);
      }
#pragma unroll
      for (int ni = 0; ni < 2; ++ni) {          // B: 16 f32 -> 16 bf16
        const f32x4* p = (const f32x4*)(wrow[ni] + kk);
        f32x4 f0 = p[0], f1 = p[1], f2 = p[2], f3 = p[3];
        b[ni].u[0] = pk_bf16(f0.x, f0.y);
        b[ni].u[1] = pk_bf16(f0.z, f0.w);
        b[ni].u[2] = pk_bf16(f1.x, f1.y);
        b[ni].u[3] = pk_bf16(f1.z, f1.w);
        b[ni].u[4] = pk_bf16(f2.x, f2.y);
        b[ni].u[5] = pk_bf16(f2.z, f2.w);
        b[ni].u[6] = pk_bf16(f3.x, f3.y);
        b[ni].u[7] = pk_bf16(f3.z, f3.w);
      }
#pragma unroll
      for (int mi = 0; mi < 2; ++mi)
#pragma unroll
        for (int ni = 0; ni < 2; ++ni)
          acc[mi][ni] = __builtin_amdgcn_wmma_f32_16x16x32_bf16(
              false, a[mi].v, false, b[ni].v, (short)0, acc[mi][ni],
              false, false);
    }
    __syncthreads();
  }

  // Epilogue: bias + relu + bf16 store. D layout: VGPR r -> row r + lhi*8.
#pragma unroll
  for (int mi = 0; mi < 2; ++mi) {
#pragma unroll
    for (int ni = 0; ni < 2; ++ni) {
      const int nn = n0 + ni * 16 + l15;
      const float bv = bias[nn];
#pragma unroll
      for (int r = 0; r < 8; ++r) {
        const int mm = mBlk + mLoc + mi * 16 + lhi * 8 + r;
        float v = acc[mi][ni][r] + bv;
        v = v < 0.0f ? 0.0f : v;
        H[(size_t)mm * N + nn] = bf16_rne(v);
      }
    }
  }
}

// ---------------------------------------------------------------------------
// Kernel 4/5: small heads Out = A @ W^T + b, f32 out, N not multiple of 16.
// One 16x16 WMMA tile per wave; weight rows clamped, stores guarded.
// Tile index is wave-uniform so EXEC is all-ones at every WMMA.
// ---------------------------------------------------------------------------
__global__ void head_wmma_kernel(const uint16_t* __restrict__ A,
                                 const float* __restrict__ W,
                                 const float* __restrict__ bias,
                                 float* __restrict__ Out, int N) {
  const int K = 4096;
  const int tiles_n = (N + 15) >> 4;
  const int tile = blockIdx.x * (blockDim.x >> 5) + (threadIdx.x >> 5);
  if (tile >= 16 * tiles_n) return;            // wave-uniform

  const int tm = tile / tiles_n;
  const int tn = tile % tiles_n;
  const int lane = threadIdx.x & 31;
  const int l15  = lane & 15;
  const int lhi  = lane >> 4;

  int nrow = tn * 16 + l15;
  if (nrow > N - 1) nrow = N - 1;              // clamp, result discarded

  const uint16_t* arow = A + (size_t)(tm * 16 + l15) * K + lhi * 16;
  const float*    wrow = W + (size_t)nrow * K + lhi * 16;

  v8f acc = {};
  for (int k = 0; k < K; k += 32) {
    Frag16 a, b;
    a.q[0] = *(const u32x4*)(arow + k);
    a.q[1] = *(const u32x4*)(arow + k + 8);
    const f32x4* p = (const f32x4*)(wrow + k);
    f32x4 f0 = p[0], f1 = p[1], f2 = p[2], f3 = p[3];
    b.u[0] = pk_bf16(f0.x, f0.y);
    b.u[1] = pk_bf16(f0.z, f0.w);
    b.u[2] = pk_bf16(f1.x, f1.y);
    b.u[3] = pk_bf16(f1.z, f1.w);
    b.u[4] = pk_bf16(f2.x, f2.y);
    b.u[5] = pk_bf16(f2.z, f2.w);
    b.u[6] = pk_bf16(f3.x, f3.y);
    b.u[7] = pk_bf16(f3.z, f3.w);
    acc = __builtin_amdgcn_wmma_f32_16x16x32_bf16(
        false, a.v, false, b.v, (short)0, acc, false, false);
  }

  const int nn = tn * 16 + l15;
  if (nn < N) {
    const float bv = bias[nn];
#pragma unroll
    for (int r = 0; r < 8; ++r) {
      const int mm = tm * 16 + lhi * 8 + r;
      Out[(size_t)mm * N + nn] = acc[r] + bv;
    }
  }
}

// ---------------------------------------------------------------------------
// Launch: roi_pool -> FC1 -> FC2 -> cls head -> regr head (all on `stream`).
// Workspace: A1 bf16 [256x25088] (12.85 MB), H1/H2 bf16 [256x4096] (2 MB ea).
// ---------------------------------------------------------------------------
extern "C" void kernel_launch(void* const* d_in, const int* in_sizes, int n_in,
                              void* d_out, int out_size, void* d_ws, size_t ws_size,
                              hipStream_t stream) {
  const float* feat  = (const float*)d_in[0];
  const float* boxes = (const float*)d_in[1];
  const float* W1    = (const float*)d_in[2];
  const float* b1    = (const float*)d_in[3];
  const float* W2    = (const float*)d_in[4];
  const float* b2    = (const float*)d_in[5];
  const float* Wc    = (const float*)d_in[6];
  const float* bc    = (const float*)d_in[7];
  const float* Wr    = (const float*)d_in[8];
  const float* br    = (const float*)d_in[9];

  uint16_t* A1 = (uint16_t*)d_ws;                 // 256*25088 bf16
  uint16_t* H1 = A1 + (size_t)256 * 25088;        // 256*4096 bf16
  uint16_t* H2 = H1 + (size_t)256 * 4096;         // 256*4096 bf16

  roi_pool_kernel<<<256, 256, 0, stream>>>(feat, boxes, A1);

  dim3 g1(4096 / 128, 256 / 64);                  // (32, 4) blocks
  gemm_relu_bf16_kernel<<<g1, 256, 0, stream>>>(A1, W1, b1, H1, 25088, 4096);
  gemm_relu_bf16_kernel<<<g1, 256, 0, stream>>>(H1, W2, b2, H2, 4096, 4096);

  float* out_cls = (float*)d_out;                 // [256, 21]
  float* out_reg = out_cls + (size_t)256 * 21;    // [256, 84]
  head_wmma_kernel<<<4, 256, 0, stream>>>(H2, Wc, bc, out_cls, 21);   // 32 tiles
  head_wmma_kernel<<<12, 256, 0, stream>>>(H2, Wr, br, out_reg, 84);  // 96 tiles
}
